// GraphWaveNetModel_66391604462226
// MI455X (gfx1250) — matrix operand order repaired and checked by reading, not compiled, not run on hardware
//
#include <hip/hip_runtime.h>
#include <math.h>

// ---------------- model constants ----------------
#define BB 8
#define TT 12
#define NN 1024
#define HH 64
#define FFdim 256
#define EMB 16
#define EE 32768
#define LL 4
#define HORIZON 12

typedef __attribute__((ext_vector_type(16))) _Float16 v16h;
typedef __attribute__((ext_vector_type(8)))  float    v8f;

enum { GF_ACC = 1, GF_BIAS = 2, GF_RELU = 4 };

// =====================================================================
// Generic WMMA GEMM:  C[r, colBase+c] (+)= sum_k A[r, k] * W[k, colBase+c]
//   - 64x64 output tile per block, 128 threads = 4 wave32 waves
//   - A rows mapped via (rowsPerBatchA, batchStrideA, ldA) to support
//     time-shifted views; blockIdx.z batches W/C (adjacency apply).
// =====================================================================
__global__ __launch_bounds__(128) void wmma_gemm_kernel(
    const float* __restrict__ A, int rowsPerBatchA, long batchStrideA, int ldA,
    const float* __restrict__ W, long batchStrideW, int ldW,
    const float* __restrict__ bias,
    float* __restrict__ C, long batchStrideC, int ldC,
    int K, int flags)
{
    const int tid  = threadIdx.x;
    const int wave = tid >> 5;
    const int lane = tid & 31;
    const int rowBase = blockIdx.x * 64;
    const int colBase = blockIdx.y * 64;
    const int z = blockIdx.z;

    const float* Wz = W + (long)z * batchStrideW;
    float*       Cz = C + (long)z * batchStrideC;

    __shared__ _Float16 As[64][40];   // [row][k]     (padded)
    __shared__ _Float16 Bs[64][40];   // [col][k]     (W transposed)

    v8f acc[4] = {};

    // A staging: thread loads 16 consecutive k of one row
    const int arow = tid >> 1;
    const int akh  = (tid & 1) * 16;
    const int gr   = rowBase + arow;
    const int ab   = gr / rowsPerBatchA;
    const int arr  = gr - ab * rowsPerBatchA;
    const float* Arow = A + (long)ab * batchStrideA + (long)arr * ldA + akh;

    // B staging: thread loads 16 consecutive cols of one k-row of W
    const int bkk = tid >> 2;          // 0..31
    const int bcs = (tid & 3) * 16;    // 0,16,32,48
    const float* Wrow = Wz + (long)bkk * ldW + colBase + bcs;

    for (int k0 = 0; k0 < K; k0 += 32) {
        {
            const float* src = Arow + k0;
            #pragma unroll
            for (int j = 0; j < 16; ++j) As[arow][akh + j] = (_Float16)src[j];
            if (k0 + 32 < K) __builtin_prefetch(src + 32, 0, 1);
        }
        {
            const float* src = Wrow + (long)k0 * ldW;
            #pragma unroll
            for (int j = 0; j < 16; ++j) Bs[bcs + j][bkk] = (_Float16)src[j];
            if (k0 + 32 < K) __builtin_prefetch(src + (long)32 * ldW, 0, 1);
        }
        __syncthreads();

        // A fragment (16x32 f16): lanes<16 -> K 0..7 & 16..23 ; lanes>=16 -> K 8..15 & 24..31
        v16h afrag;
        {
            const _Float16* ap = &As[(wave << 4) + (lane & 15)][(lane < 16) ? 0 : 8];
            #pragma unroll
            for (int j = 0; j < 8; ++j) { afrag[j] = ap[j]; afrag[j + 8] = ap[j + 16]; }
        }
        // B fragments (32x16 f16): lanes 0-15 hold K 0..7 & 16..23 of col n, lanes 16-31 hold K 8..15 & 24..31
        const int bk = (lane < 16) ? 0 : 8;
        const int bc = lane & 15;
        #pragma unroll
        for (int nt = 0; nt < 4; ++nt) {
            const _Float16* bp = &Bs[nt * 16 + bc][bk];
            v16h bf;
            #pragma unroll
            for (int j = 0; j < 8; ++j) { bf[j] = bp[j]; bf[j + 8] = bp[j + 16]; }
            acc[nt] = __builtin_amdgcn_wmma_f32_16x16x32_f16(
                false, afrag, false, bf, (short)0, acc[nt], false, false);
        }
        __syncthreads();
    }

    // D layout: VGPR j -> M=j (lanes<16) / M=8+j (lanes>=16), N = lane&15
    const int om = (lane < 16) ? 0 : 8;
    const int on = lane & 15;
    const int orow0 = rowBase + (wave << 4) + om;
    #pragma unroll
    for (int nt = 0; nt < 4; ++nt) {
        const int col = colBase + nt * 16 + on;
        const float bv = (flags & GF_BIAS) ? bias[col] : 0.0f;
        #pragma unroll
        for (int j = 0; j < 8; ++j) {
            const int row = orow0 + j;
            const long off = (long)row * ldC + col;
            float val = acc[nt][j] + bv;
            if (flags & GF_ACC)  val += Cz[off];
            if (flags & GF_RELU) val = fmaxf(val, 0.0f);
            Cz[off] = val;
        }
    }
}

// ---------------- small helper kernels ----------------
__global__ void zero_kernel(float* p, long n) {
    long i = (long)blockIdx.x * 256 + threadIdx.x;
    if (i < n) p[i] = 0.0f;
}

__global__ void encoder_kernel(const float* __restrict__ x,
                               const float* __restrict__ W,
                               const float* __restrict__ b,
                               float* __restrict__ h, long rows) {
    long i = (long)blockIdx.x * 256 + threadIdx.x;
    if (i >= rows * HH) return;
    int c = (int)(i & (HH - 1));
    long r = i >> 6;
    h[i] = x[r] * W[c] + b[c];
}

// adj = softmax(relu(src_emb @ tgt_emb^T), axis=1); one block per row
__global__ __launch_bounds__(256) void adj_row_kernel(
    const float* __restrict__ se, const float* __restrict__ te,
    float* __restrict__ adj) {
    const int w = blockIdx.x, tid = threadIdx.x;
    __shared__ float sv[EMB];
    __shared__ float red[256];
    if (tid < EMB) sv[tid] = se[w * EMB + tid];
    __syncthreads();
    float vals[4]; float mx = 0.0f;
    #pragma unroll
    for (int q = 0; q < 4; ++q) {
        int v = tid + q * 256;
        const float* tr = te + (long)v * EMB;
        float dot = 0.0f;
        #pragma unroll
        for (int c = 0; c < EMB; ++c) dot += sv[c] * tr[c];
        dot = fmaxf(dot, 0.0f);
        vals[q] = dot; mx = fmaxf(mx, dot);
    }
    red[tid] = mx; __syncthreads();
    for (int s = 128; s > 0; s >>= 1) { if (tid < s) red[tid] = fmaxf(red[tid], red[tid + s]); __syncthreads(); }
    mx = red[0]; __syncthreads();
    float sum = 0.0f;
    #pragma unroll
    for (int q = 0; q < 4; ++q) { vals[q] = expf(vals[q] - mx); sum += vals[q]; }
    red[tid] = sum; __syncthreads();
    for (int s = 128; s > 0; s >>= 1) { if (tid < s) red[tid] += red[tid + s]; __syncthreads(); }
    const float inv = 1.0f / red[0];
    #pragma unroll
    for (int q = 0; q < 4; ++q) adj[(long)w * NN + tid + q * 256] = vals[q] * inv;
}

__global__ void deg_kernel(const int* __restrict__ src, const int* __restrict__ dst,
                           const float* __restrict__ ew,
                           float* __restrict__ degf, float* __restrict__ degb) {
    int e = blockIdx.x * 256 + threadIdx.x;
    if (e >= EE) return;
    atomicAdd(&degf[dst[e]], ew[e]);
    atomicAdd(&degb[src[e]], ew[e]);
}

__global__ void edgew_kernel(const int* __restrict__ src, const int* __restrict__ dst,
                             const float* __restrict__ ew,
                             const float* __restrict__ degf, const float* __restrict__ degb,
                             float* __restrict__ wf, float* __restrict__ wb) {
    int e = blockIdx.x * 256 + threadIdx.x;
    if (e >= EE) return;
    float df = degf[dst[e]]; wf[e] = ew[e] / (df > 0.0f ? df : 1.0f);
    float dbv = degb[src[e]]; wb[e] = ew[e] / (dbv > 0.0f ? dbv : 1.0f);
}

// scatter-add diffusion: out[bt, d, c] += h[bt, s, c] * w[e]
__global__ __launch_bounds__(64) void prop_kernel(
    const float* __restrict__ hin, const int* __restrict__ s_idx,
    const int* __restrict__ d_idx, const float* __restrict__ w,
    float* __restrict__ out) {
    const int e = blockIdx.x, bt = blockIdx.y, c = threadIdx.x;
    const int sN = s_idx[e], dN = d_idx[e];
    const float val = hin[((long)bt * NN + sN) * HH + c] * w[e];
    atomicAdd(&out[((long)bt * NN + dN) * HH + c], val);
}

__global__ void gate_kernel(float* __restrict__ f, const float* __restrict__ g, long n) {
    long i = (long)blockIdx.x * 256 + threadIdx.x;
    if (i >= n) return;
    const float fv = f[i], gv = g[i];
    f[i] = tanhf(fv) * (1.0f / (1.0f + expf(-gv)));
}

__global__ void resadd_kernel(float* __restrict__ hn, const float* __restrict__ hc,
                              int To, int Tin, int doff, long total) {
    long i = (long)blockIdx.x * 256 + threadIdx.x;
    if (i >= total) return;
    const long perB = (long)To * NN * HH;
    const int b = (int)(i / perB);
    const long rem = i - (long)b * perB;
    hn[i] += hc[(long)b * Tin * NN * HH + (long)doff * NN * HH + rem];
}

// training-mode batchnorm over (B,T,N) per channel; one block per channel
__global__ __launch_bounds__(256) void bn_kernel(float* __restrict__ h, long R,
                                                 const float* __restrict__ gamma,
                                                 const float* __restrict__ beta) {
    const int c = blockIdx.x, tid = threadIdx.x;
    float s = 0.0f, s2 = 0.0f;
    for (long r = tid; r < R; r += 256) { float v = h[r * HH + c]; s += v; s2 += v * v; }
    __shared__ float sh[256], sh2[256];
    sh[tid] = s; sh2[tid] = s2; __syncthreads();
    for (int st = 128; st > 0; st >>= 1) {
        if (tid < st) { sh[tid] += sh[tid + st]; sh2[tid] += sh2[tid + st]; }
        __syncthreads();
    }
    const float m = sh[0] / (float)R;
    const float var = sh2[0] / (float)R - m * m;
    const float inv = rsqrtf(var + 1e-5f);
    const float ga = gamma[c], be = beta[c];
    for (long r = tid; r < R; r += 256) { long o = r * HH + c; h[o] = (h[o] - m) * inv * ga + be; }
}

__global__ void relu_kernel(float* __restrict__ p, long n) {
    long i = (long)blockIdx.x * 256 + threadIdx.x;
    if (i < n) p[i] = fmaxf(p[i], 0.0f);
}

// final projection 512 -> HORIZON, with transpose to [B, HORIZON, N, 1]
__global__ void dec2_kernel(const float* __restrict__ D1, const float* __restrict__ W2,
                            const float* __restrict__ b2, float* __restrict__ out) {
    int gid = blockIdx.x * 256 + threadIdx.x;
    if (gid >= BB * NN * HORIZON) return;
    const int hor = gid % HORIZON;
    const int bn  = gid / HORIZON;
    const float* row = D1 + (long)bn * (2 * FFdim);
    float s = b2[hor];
    for (int k = 0; k < 2 * FFdim; ++k) s += row[k] * W2[k * HORIZON + hor];
    const int b = bn >> 10, n = bn & (NN - 1);
    out[((long)b * HORIZON + hor) * NN + n] = s;
}

// ---------------- host-side GEMM launcher ----------------
static inline void gemm(hipStream_t s,
                        const float* A, int rpbA, long bsA, int ldA,
                        const float* W, long bsW, int ldW, const float* bias,
                        float* C, long bsC, int ldC,
                        int R, int K, int Nout, int nBatch, int flags) {
    dim3 g(R / 64, Nout / 64, nBatch);
    wmma_gemm_kernel<<<g, 128, 0, s>>>(A, rpbA, bsA, ldA, W, bsW, ldW, bias,
                                       C, bsC, ldC, K, flags);
}

static inline void zero(hipStream_t s, float* p, long n) {
    zero_kernel<<<(int)((n + 255) / 256), 256, 0, s>>>(p, n);
}

extern "C" void kernel_launch(void* const* d_in, const int* in_sizes, int n_in,
                              void* d_out, int out_size, void* d_ws, size_t ws_size,
                              hipStream_t stream) {
    (void)in_sizes; (void)n_in; (void)out_size; (void)ws_size;

    const float* x       = (const float*)d_in[0];
    const int*   eidx    = (const int*)d_in[1];
    const float* ew      = (const float*)d_in[2];
    const float* src_emb = (const float*)d_in[3];
    const float* tgt_emb = (const float*)d_in[4];
    const float* enc_W   = (const float*)d_in[5];
    const float* enc_b   = (const float*)d_in[6];
    const float* tWf     = (const float*)d_in[7];
    const float* tbf     = (const float*)d_in[8];
    const float* tWg     = (const float*)d_in[9];
    const float* tbg     = (const float*)d_in[10];
    const float* dfW     = (const float*)d_in[11];
    const float* dfb     = (const float*)d_in[12];
    const float* deW     = (const float*)d_in[13];
    const float* deb     = (const float*)d_in[14];
    const float* skW     = (const float*)d_in[15];
    const float* skb     = (const float*)d_in[16];
    const float* gam     = (const float*)d_in[17];
    const float* bet     = (const float*)d_in[18];
    const float* W1      = (const float*)d_in[19];
    const float* b1      = (const float*)d_in[20];
    const float* W2      = (const float*)d_in[21];
    const float* b2      = (const float*)d_in[22];
    const int* esrc = eidx;
    const int* edst = eidx + EE;

    // workspace layout (floats)
    float* w = (float*)d_ws;
    const long HMAX = (long)BB * TT * NN * HH;
    float* hb0 = w; w += HMAX;
    float* hb1 = w; w += HMAX;
    float* fb  = w; w += HMAX;           // f -> gated h
    float* gb  = w; w += HMAX;           // g
    float* P0  = w; w += HMAX;           // prop / dense scratch
    float* P1  = w; w += HMAX;
    float* adj = w; w += (long)NN * NN;
    float* S   = w; w += (long)BB * NN * FFdim;     // skip accum (last step)
    float* D1  = w; w += (long)BB * NN * 2 * FFdim; // decoder hidden
    float* wf  = w; w += EE;
    float* wb  = w; w += EE;
    float* dgf = w; w += NN;
    float* dgb = w; w += NN;

    // ---- setup ----
    {
        const long rows = (long)BB * TT * NN;
        encoder_kernel<<<(int)((rows * HH + 255) / 256), 256, 0, stream>>>(x, enc_W, enc_b, hb0, rows);
    }
    zero(stream, dgf, 2 * NN);  // dgf and dgb are contiguous
    deg_kernel<<<EE / 256, 256, 0, stream>>>(esrc, edst, ew, dgf, dgb);
    edgew_kernel<<<EE / 256, 256, 0, stream>>>(esrc, edst, ew, dgf, dgb, wf, wb);
    adj_row_kernel<<<NN, 256, 0, stream>>>(src_emb, tgt_emb, adj);

    float* hcur = hb0;
    float* hnx  = hb1;
    int Tin = TT;

    for (int i = 0; i < LL; ++i) {
        const int d  = (i % 2 == 0) ? 1 : 2;   // DILATIONS = [1,2,1,2]
        const int To = Tin - d;
        const int BT = BB * To;
        const int R  = BT * NN;
        const long nel = (long)R * HH;
        const int rpbT = To * NN;
        const long bsT = (long)Tin * NN * HH;

        // ---- gated dilated temporal conv (K=2 taps) ----
        gemm(stream, hcur,                   rpbT, bsT, HH, tWf + (long)(i * 2 + 0) * HH * HH, 0, HH, tbf + i * HH, fb, 0, HH, R, HH, HH, 1, GF_BIAS);
        gemm(stream, hcur + (long)d * NN * HH, rpbT, bsT, HH, tWf + (long)(i * 2 + 1) * HH * HH, 0, HH, nullptr,      fb, 0, HH, R, HH, HH, 1, GF_ACC);
        gemm(stream, hcur,                   rpbT, bsT, HH, tWg + (long)(i * 2 + 0) * HH * HH, 0, HH, tbg + i * HH, gb, 0, HH, R, HH, HH, 1, GF_BIAS);
        gemm(stream, hcur + (long)d * NN * HH, rpbT, bsT, HH, tWg + (long)(i * 2 + 1) * HH * HH, 0, HH, nullptr,      gb, 0, HH, R, HH, HH, 1, GF_ACC);
        gate_kernel<<<(int)((nel + 255) / 256), 256, 0, stream>>>(fb, gb, nel);
        float* hnew = fb;

        // ---- skip connection (only last timestep feeds the readout) ----
        gemm(stream, hnew + (long)(To - 1) * NN * HH, NN, (long)To * NN * HH, HH,
             skW + (long)i * HH * FFdim, 0, FFdim, skb + i * FFdim,
             S, 0, FFdim, BB * NN, HH, FFdim, 1, GF_BIAS | (i ? GF_ACC : 0));

        // ---- K-order sparse diffusion conv ----
        const float* dWi = dfW + (long)i * 320 * HH;
        gemm(stream, hnew, R, 0, HH, dWi + 0 * HH * HH, 0, HH, dfb + i * HH, hnx, 0, HH, R, HH, HH, 1, GF_BIAS);

        dim3 pg(EE, BT);
        zero(stream, P0, nel);
        prop_kernel<<<pg, HH, 0, stream>>>(hnew, esrc, edst, wf, P0);
        gemm(stream, P0, R, 0, HH, dWi + 1 * HH * HH, 0, HH, nullptr, hnx, 0, HH, R, HH, HH, 1, GF_ACC);
        zero(stream, P1, nel);
        prop_kernel<<<pg, HH, 0, stream>>>(P0, esrc, edst, wf, P1);
        gemm(stream, P1, R, 0, HH, dWi + 2 * HH * HH, 0, HH, nullptr, hnx, 0, HH, R, HH, HH, 1, GF_ACC);
        zero(stream, P0, nel);
        prop_kernel<<<pg, HH, 0, stream>>>(hnew, edst, esrc, wb, P0);
        gemm(stream, P0, R, 0, HH, dWi + 3 * HH * HH, 0, HH, nullptr, hnx, 0, HH, R, HH, HH, 1, GF_ACC);
        zero(stream, P1, nel);
        prop_kernel<<<pg, HH, 0, stream>>>(P0, edst, esrc, wb, P1);
        gemm(stream, P1, R, 0, HH, dWi + 4 * HH * HH, 0, HH, nullptr, hnx, 0, HH, R, HH, HH, 1, GF_ACC);

        // ---- dense learned-adjacency conv (dominant GEMMs, batched over b,t) ----
        gemm(stream, adj, NN, 0, NN, hnew, (long)NN * HH, HH, nullptr, P0, (long)NN * HH, HH, NN, NN, HH, BT, 0);
        gemm(stream, adj, NN, 0, NN, P0,   (long)NN * HH, HH, nullptr, P1, (long)NN * HH, HH, NN, NN, HH, BT, 0);
        const float* eWi = deW + (long)i * 128 * HH;
        gemm(stream, P0, R, 0, HH, eWi + 0 * HH * HH, 0, HH, deb + i * HH, hnx, 0, HH, R, HH, HH, 1, GF_ACC | GF_BIAS);
        gemm(stream, P1, R, 0, HH, eWi + 1 * HH * HH, 0, HH, nullptr,      hnx, 0, HH, R, HH, HH, 1, GF_ACC);

        // ---- residual + batchnorm ----
        resadd_kernel<<<(int)((nel + 255) / 256), 256, 0, stream>>>(hnx, hcur, To, Tin, d, nel);
        bn_kernel<<<HH, 256, 0, stream>>>(hnx, (long)R, gam + i * HH, bet + i * HH);

        float* tmp = hcur; hcur = hnx; hnx = tmp;
        Tin = To;
    }

    // ---- readout MLP ----
    relu_kernel<<<(int)(((long)BB * NN * FFdim + 255) / 256), 256, 0, stream>>>(S, (long)BB * NN * FFdim);
    gemm(stream, S, BB * NN, 0, FFdim, W1, 0, 2 * FFdim, b1, D1, 0, 2 * FFdim,
         BB * NN, FFdim, 2 * FFdim, 1, GF_BIAS | GF_RELU);
    dec2_kernel<<<(BB * NN * HORIZON + 255) / 256, 256, 0, stream>>>(D1, W2, b2, (float*)d_out);
}
